// LlamaAttention_72902774882862
// MI455X (gfx1250) — matrix operand order repaired and checked by reading, not compiled, not run on hardware
//
#include <hip/hip_runtime.h>
#include <hip/hip_bf16.h>
#include <math.h>

typedef __bf16 bf16_t;
typedef __attribute__((ext_vector_type(16))) __bf16 v16bf;
typedef __attribute__((ext_vector_type(8)))  float  v8f;

constexpr int Bc = 2, Lc = 2048, Dc = 2048, Hc = 16, HDc = 128;
constexpr int Mrows = Bc * Lc;  // 4096
constexpr int PAD32 = 40;       // padded LDS stride for 32-wide tiles (80B, 16B-mult)
constexpr int PAD128 = 136;     // padded LDS stride for 128-wide tiles (272B)

__device__ inline v8f v8f_zero() { v8f z = {}; return z; }

__device__ inline v8f wmma_bf16f32(v16bf a, v16bf b, v8f c) {
  return __builtin_amdgcn_wmma_f32_16x16x32_bf16(false, a, false, b, (short)0, c,
                                                 false, false);
}

// A fragment: 16x32 tile, row-major source (stride ld elements).
// lanes 0-15: row M=lane, K {0..7,16..23}; lanes 16-31: K {8..15,24..31}.
// Per-lane: two contiguous 16B reads -> 2x ds_load_b128 / global_load_b128.
__device__ inline v16bf frag_a_load(const bf16_t* p, int ld) {
  const int lane = threadIdx.x & 31;
  const bf16_t* src = p + (lane & 15) * ld + ((lane >> 4) << 3);
  v16bf a;
#pragma unroll
  for (int i = 0; i < 8; ++i) a[i] = src[i];
#pragma unroll
  for (int i = 0; i < 8; ++i) a[i + 8] = src[i + 16];
  return a;
}

// B fragment from TRANSPOSED storage: Mem[n][k] row-major (stride ld elements).
// B[k][n] = Mem[n][k]; per-lane K {0..7,16..23} / {8..15,24..31} -> 2x ds_load_b128.
__device__ inline v16bf frag_bt_load(const bf16_t* p, int ld) {
  const int lane = threadIdx.x & 31;
  const bf16_t* src = p + (lane & 15) * ld + ((lane >> 4) << 3);
  v16bf b;
#pragma unroll
  for (int i = 0; i < 8; ++i) b[i] = src[i];
#pragma unroll
  for (int i = 0; i < 8; ++i) b[i + 8] = src[i + 16];
  return b;
}

__global__ __launch_bounds__(256)
void cvt_f32_bf16(const float* __restrict__ in, bf16_t* __restrict__ out, int n) {
  int i = blockIdx.x * blockDim.x + threadIdx.x;
  if (i < n) out[i] = (bf16_t)in[i];
}

// Transpose + convert: in = K x N f32 row-major, out = N x K bf16 row-major.
__global__ __launch_bounds__(256)
void cvt_transpose_kernel(const float* __restrict__ in, bf16_t* __restrict__ out) {
  __shared__ float tile[16][17];
  const int tx = threadIdx.x & 15, ty = threadIdx.x >> 4;
  const int k0 = blockIdx.x * 16, n0 = blockIdx.y * 16;
  tile[ty][tx] = in[(size_t)(k0 + ty) * Dc + n0 + tx];
  __syncthreads();
  out[(size_t)(n0 + ty) * Dc + k0 + tx] = (bf16_t)tile[tx][ty];
}

// Traditional RoPE on interleaved pairs within each head slice; scale folded in.
__global__ __launch_bounds__(256)
void rope_kernel(bf16_t* __restrict__ t, float scale, int npairs) {
  int p = blockIdx.x * blockDim.x + threadIdx.x;
  if (p >= npairs) return;
  const long long e = (long long)p * 2;
  const int d  = (int)(e % Dc);
  const int l  = (int)((e / Dc) % Lc);
  const int fi = (d % HDc) >> 1;
  const float freq = __expf(-(float)fi * (9.210340371976184f / 64.0f)); // ln(1e4)/64
  float sn, cs;
  __sincosf((float)l * freq, &sn, &cs);
  const float x1 = (float)t[e];
  const float x2 = (float)t[e + 1];
  t[e]     = (bf16_t)((x1 * cs - x2 * sn) * scale);
  t[e + 1] = (bf16_t)((x1 * sn + x2 * cs) * scale);
}

// C[M,N] = A[M,K] * B[K,N], B supplied TRANSPOSED (Bt = N x K row-major).
// Block: 256 threads = 8 waves (2x4); block tile 32x512; wave tile 16x128.
// store_mode: 0 = bf16 row-major, 1 = f32 row-major, 2 = bf16 transposed (N x M).
__global__ __launch_bounds__(256)
void gemm_bf16_kernel(const bf16_t* __restrict__ A, const bf16_t* __restrict__ Bt,
                      void* __restrict__ Cout, int M, int N, int K, int store_mode) {
  __shared__ bf16_t lA[32 * PAD32];
  __shared__ bf16_t lB[512 * PAD32];
  const int tid  = threadIdx.x;
  const int bm   = blockIdx.x * 32;
  const int bn   = blockIdx.y * 512;
  const int wid  = tid >> 5;
  const int wrow = wid >> 2;  // 0..1
  const int wcol = wid & 3;   // 0..3

  v8f acc[8];
#pragma unroll
  for (int j = 0; j < 8; ++j) acc[j] = v8f_zero();

  const int ar = tid >> 3, ac = (tid & 7) * 4;  // A tile 32x32: 4 elems/thread

  for (int kk = 0; kk < K; kk += 32) {
    // A tile: 32 rows x 32 k
    const bf16_t* ga = A + (size_t)(bm + ar) * K + kk + ac;
    *(uint2*)(lA + ar * PAD32 + ac) = *(const uint2*)ga;
    // B tile: 512 n-rows x 32 k, contiguous rows of Bt
#pragma unroll
    for (int rr = 0; rr < 2; ++rr) {
      const int row = tid * 2 + rr;
      const uint4* sb = (const uint4*)(Bt + (size_t)(bn + row) * K + kk);
      uint4* db = (uint4*)(lB + row * PAD32);
#pragma unroll
      for (int i = 0; i < 4; ++i) db[i] = sb[i];
    }
    __syncthreads();

    v16bf af = frag_a_load(lA + wrow * 16 * PAD32, PAD32);
#pragma unroll
    for (int j = 0; j < 8; ++j) {
      v16bf bf = frag_bt_load(lB + (wcol * 128 + j * 16) * PAD32, PAD32);
      acc[j] = wmma_bf16f32(af, bf, acc[j]);
    }
    __syncthreads();
  }

  const int lane  = tid & 31;
  const int nn    = lane & 15;
  const int rbase = (lane >> 4) << 3;
  const int row0  = bm + wrow * 16 + rbase;
#pragma unroll
  for (int j = 0; j < 8; ++j) {
    const int col = bn + wcol * 128 + j * 16 + nn;
    if (store_mode == 1) {
      float* C = (float*)Cout;
#pragma unroll
      for (int i = 0; i < 8; ++i) C[(size_t)(row0 + i) * N + col] = acc[j][i];
    } else if (store_mode == 0) {
      bf16_t* C = (bf16_t*)Cout;
#pragma unroll
      for (int i = 0; i < 8; ++i) C[(size_t)(row0 + i) * N + col] = (bf16_t)acc[j][i];
    } else {  // transposed: Ct[col][row], contiguous over i
      bf16_t* C = (bf16_t*)Cout;
#pragma unroll
      for (int i = 0; i < 8; ++i) C[(size_t)col * M + row0 + i] = (bf16_t)acc[j][i];
    }
  }
}

// Flash attention. Q,K: (B,L,D) bf16 (RoPE'd, Q pre-scaled). Vt: (D, B*L) bf16.
// Block = (b,h, 64-query chunk), 4 waves x 16 queries.
__global__ __launch_bounds__(128)
void attn_kernel(const bf16_t* __restrict__ Q, const bf16_t* __restrict__ Km,
                 const bf16_t* __restrict__ Vt, bf16_t* __restrict__ O) {
  __shared__ bf16_t lK[32 * PAD128];   // [key][hd]
  __shared__ bf16_t lV[128 * PAD32];   // [hd][key] (transposed)
  __shared__ bf16_t lP[4 * 16 * PAD32];

  const int bh    = blockIdx.x;
  const int b     = bh / Hc, h = bh % Hc;
  const int qblk  = blockIdx.y * 64;
  const int tid   = threadIdx.x;
  const int w     = tid >> 5;
  const int lane  = tid & 31;
  const int nn    = lane & 15;
  const int rbase = (lane >> 4) << 3;
  const int q0    = qblk + w * 16;

  const size_t headoff  = ((size_t)b * Lc) * Dc + (size_t)h * HDc;   // Q,K,O
  const size_t vheadoff = ((size_t)h * HDc) * Mrows + (size_t)b * Lc; // Vt rows

  const bf16_t* Qbase = Q + headoff + (size_t)q0 * Dc;
  v16bf qf[4];
#pragma unroll
  for (int kc = 0; kc < 4; ++kc) qf[kc] = frag_a_load(Qbase + kc * 32, Dc);

  v8f acc[8];
#pragma unroll
  for (int j = 0; j < 8; ++j) acc[j] = v8f_zero();
  float mrow[8], lrow[8];
#pragma unroll
  for (int i = 0; i < 8; ++i) { mrow[i] = -1e30f; lrow[i] = 0.f; }

  const int kr = tid >> 2;        // 0..31 key row (K tile)
  const int kc4 = (tid & 3) * 32; // hd chunk

  const int nkt = (qblk + 64) / 32;
  for (int kt = 0; kt < nkt; ++kt) {
    const int key0 = kt * 32;
    __syncthreads();
    {
      // K tile: 32 keys x 128 hd, row-major source
      const uint4* sk = (const uint4*)(Km + headoff + (size_t)(key0 + kr) * Dc + kc4);
      uint4* dk = (uint4*)(lK + kr * PAD128 + kc4);
#pragma unroll
      for (int i = 0; i < 4; ++i) dk[i] = sk[i];
      // V tile: 128 hd-rows x 32 keys, contiguous rows of Vt
      const uint4* sv = (const uint4*)(Vt + vheadoff + (size_t)tid * Mrows + key0);
      uint4* dv = (uint4*)(lV + tid * PAD32);
#pragma unroll
      for (int i = 0; i < 4; ++i) dv[i] = sv[i];
    }
    __syncthreads();

    if (key0 <= q0 + 15) {  // causal: any unmasked key in this tile?
      v8f s0 = v8f_zero(), s1 = v8f_zero();
#pragma unroll
      for (int kc = 0; kc < 4; ++kc) {  // HD=128 in 4 k-steps of 32
        s0 = wmma_bf16f32(qf[kc], frag_bt_load(lK + kc * 32, PAD128), s0);
        s1 = wmma_bf16f32(qf[kc], frag_bt_load(lK + 16 * PAD128 + kc * 32, PAD128), s1);
      }
#pragma unroll
      for (int i = 0; i < 8; ++i) {  // causal mask (reference adds -1e9)
        const int qr = q0 + rbase + i;
        if (key0 + nn > qr)      s0[i] += -1e9f;
        if (key0 + 16 + nn > qr) s1[i] += -1e9f;
      }
      // online softmax; butterfly over N stays within each 16-lane half
#pragma unroll
      for (int i = 0; i < 8; ++i) {
        float t = fmaxf(s0[i], s1[i]);
        t = fmaxf(t, __shfl_xor(t, 1));
        t = fmaxf(t, __shfl_xor(t, 2));
        t = fmaxf(t, __shfl_xor(t, 4));
        t = fmaxf(t, __shfl_xor(t, 8));
        const float mn   = fmaxf(mrow[i], t);
        const float corr = __expf(mrow[i] - mn);
        const float p0   = __expf(s0[i] - mn);
        const float p1   = __expf(s1[i] - mn);
        float rs = p0 + p1;
        rs += __shfl_xor(rs, 1);
        rs += __shfl_xor(rs, 2);
        rs += __shfl_xor(rs, 4);
        rs += __shfl_xor(rs, 8);
        lrow[i] = lrow[i] * corr + rs;
        mrow[i] = mn;
        s0[i] = p0; s1[i] = p1;
#pragma unroll
        for (int j = 0; j < 8; ++j) acc[j][i] *= corr;
      }
      // restage P (C layout) -> LDS -> A-fragment (wave-local, DS in-order)
      bf16_t* pw = lP + w * (16 * PAD32);
#pragma unroll
      for (int i = 0; i < 8; ++i) {
        pw[(rbase + i) * PAD32 + nn]      = (bf16_t)s0[i];
        pw[(rbase + i) * PAD32 + 16 + nn] = (bf16_t)s1[i];
      }
      asm volatile("" ::: "memory");
      v16bf pf = frag_a_load(pw, PAD32);
#pragma unroll
      for (int j = 0; j < 8; ++j) {  // P(16x32) x V(32x128) via transposed V tile
        v16bf vf = frag_bt_load(lV + (j * 16) * PAD32, PAD32);
        acc[j] = wmma_bf16f32(pf, vf, acc[j]);
      }
    }
  }

  bf16_t* Obase = O + headoff + (size_t)q0 * Dc;
#pragma unroll
  for (int i = 0; i < 8; ++i) {
    const float inv = 1.0f / lrow[i];
#pragma unroll
    for (int j = 0; j < 8; ++j)
      Obase[(size_t)(rbase + i) * Dc + j * 16 + nn] = (bf16_t)(acc[j][i] * inv);
  }
}

extern "C" void kernel_launch(void* const* d_in, const int* in_sizes, int n_in,
                              void* d_out, int out_size, void* d_ws, size_t ws_size,
                              hipStream_t stream) {
  const float* x  = (const float*)d_in[0];
  // d_in[1] = mask: causality applied analytically in attn_kernel
  const float* Wq = (const float*)d_in[2];
  const float* Wk = (const float*)d_in[3];
  const float* Wv = (const float*)d_in[4];
  const float* Wo = (const float*)d_in[5];
  float* out = (float*)d_out;

  const size_t XE = (size_t)Bc * Lc * Dc;  // 8388608
  const size_t WE = (size_t)Dc * Dc;       // 4194304

  bf16_t* p   = (bf16_t*)d_ws;
  bf16_t* xb  = p; p += XE;
  bf16_t* Wqt = p; p += WE;   // transposed bf16 weights (N x K)
  bf16_t* Wkt = p; p += WE;
  bf16_t* Wvt = p; p += WE;
  bf16_t* Wot = p; p += WE;
  bf16_t* Qb  = p; p += XE;
  bf16_t* Kb  = p; p += XE;
  bf16_t* Vtb = p; p += XE;   // V stored transposed: (D, B*L)
  bf16_t* AOb = p; p += XE;

  cvt_f32_bf16<<<(int)(XE / 256), 256, 0, stream>>>(x, xb, (int)XE);
  dim3 tgrid(Dc / 16, Dc / 16);
  cvt_transpose_kernel<<<tgrid, 256, 0, stream>>>(Wq, Wqt);
  cvt_transpose_kernel<<<tgrid, 256, 0, stream>>>(Wk, Wkt);
  cvt_transpose_kernel<<<tgrid, 256, 0, stream>>>(Wv, Wvt);
  cvt_transpose_kernel<<<tgrid, 256, 0, stream>>>(Wo, Wot);

  dim3 ggrid(Mrows / 32, Dc / 512);
  gemm_bf16_kernel<<<ggrid, 256, 0, stream>>>(xb, Wqt, Qb,  Mrows, Dc, Dc, 0);
  gemm_bf16_kernel<<<ggrid, 256, 0, stream>>>(xb, Wkt, Kb,  Mrows, Dc, Dc, 0);
  gemm_bf16_kernel<<<ggrid, 256, 0, stream>>>(xb, Wvt, Vtb, Mrows, Dc, Dc, 2);

  const int npairs = (int)(XE / 2);
  rope_kernel<<<npairs / 256, 256, 0, stream>>>(Qb, 0.08838834764831845f, npairs);
  rope_kernel<<<npairs / 256, 256, 0, stream>>>(Kb, 1.0f, npairs);

  attn_kernel<<<dim3(Bc * Hc, Lc / 64), 128, 0, stream>>>(Qb, Kb, Vtb, AOb);

  gemm_bf16_kernel<<<ggrid, 256, 0, stream>>>(AOb, Wot, out, Mrows, Dc, Dc, 1);
}
